// UnivNetLVCBlock_4260607558288
// MI455X (gfx1250) — compile-verified, matrix operand order checked by reading, not run on hardware
//
#include <hip/hip_runtime.h>
#include <hip/hip_bf16.h>
#include <math.h>

#define HID   32
#define TLEN  32768
#define SLEN  512
#define NB    4

typedef __attribute__((ext_vector_type(16))) __bf16 v16bf;
typedef __attribute__((ext_vector_type(8)))  float  v8f;
typedef unsigned int uq4 __attribute__((ext_vector_type(4)));

union Frag16 { v16bf v; __bf16 h[16]; uq4 q[2]; };

__device__ __forceinline__ float lrelu(float x, float s) { return x >= 0.f ? x : s * x; }

// ---------------------------------------------------------------------------
// 1) ConvTranspose1d (stride 8, K=16, pad 4) with input leaky-relu(0.2)
// ---------------------------------------------------------------------------
__global__ void k_convt(const float* __restrict__ hidden, const float* __restrict__ w,
                        const float* __restrict__ bias, float* __restrict__ out) {
  int e = blockIdx.x * blockDim.x + threadIdx.x;   // 4*32*32768
  int n = e & (TLEN - 1);
  int c = (e >> 15) & (HID - 1);
  int b = e >> 20;
  float acc = bias[c];
  int r = (n + 4) & 7;
#pragma unroll
  for (int t = 0; t < 2; ++t) {
    int k = r + 8 * t;
    int m = (n + 4 - k) >> 3;
    if (m >= 0 && m < 4096) {
      const float* xp = hidden + (size_t)(b * HID) * 4096 + m;
#pragma unroll
      for (int ci = 0; ci < HID; ++ci)
        acc += lrelu(xp[(size_t)ci * 4096], 0.2f) * w[(ci * HID + c) * 16 + k];
    }
  }
  out[e] = acc;
}

// ---------------------------------------------------------------------------
// 2) Kernel-predictor frontend (tiny): in-conv + 3 residual blocks, all f32
// ---------------------------------------------------------------------------
__global__ void k_kp_in(const float* __restrict__ spec, const float* __restrict__ w,
                        const float* __restrict__ bias, float* __restrict__ out) {
  int e = blockIdx.x * blockDim.x + threadIdx.x;   // 4*64*512
  int s = e & 511; int c = (e >> 9) & 63; int b = e >> 15;
  float acc = bias[c];
  for (int cc = 0; cc < 100; ++cc) {
    const float* xp = spec + (size_t)(b * 100 + cc) * SLEN;
    const float* wp = w + (c * 100 + cc) * 5;
#pragma unroll
    for (int kk = 0; kk < 5; ++kk) {
      int q = s + kk - 2;
      if (q >= 0 && q < SLEN) acc += xp[q] * wp[kk];
    }
  }
  out[e] = lrelu(acc, 0.1f);
}

__global__ void k_kp_res(const float* __restrict__ in, const float* __restrict__ w,
                         const float* __restrict__ bias, const float* __restrict__ resid,
                         float* __restrict__ out) {
  int e = blockIdx.x * blockDim.x + threadIdx.x;   // 4*64*512
  int s = e & 511; int c = (e >> 9) & 63; int b = e >> 15;
  float acc = bias[c];
  for (int cc = 0; cc < 64; ++cc) {
    const float* xp = in + (size_t)(b * 64 + cc) * SLEN;
    const float* wp = w + (c * 64 + cc) * 3;
#pragma unroll
    for (int kk = 0; kk < 3; ++kk) {
      int q = s + kk - 1;
      if (q >= 0 && q < SLEN) acc += xp[q] * wp[kk];
    }
  }
  float v = lrelu(acc, 0.1f);
  if (resid) v += resid[e];
  out[e] = v;
}

// ---------------------------------------------------------------------------
// 3) GEMM operand prep, bf16.
//    B im2col now [b][s][K=192] so WMMA B-frags are contiguous 16B loads.
// ---------------------------------------------------------------------------
__global__ void k_im2col(const float* __restrict__ hkp, __bf16* __restrict__ Bbf) {
  int e = blockIdx.x * blockDim.x + threadIdx.x;   // 4*512*192
  int K = e % 192; int s = (e / 192) & 511; int b = e / (192 * 512);
  int c = K / 3, kk = K - c * 3;
  int q = s + kk - 1;
  float v = (q >= 0 && q < SLEN) ? hkp[(size_t)(b * 64 + c) * SLEN + q] : 0.f;
  Bbf[e] = (__bf16)v;
}

__global__ void k_wcvt(const float* __restrict__ w, __bf16* __restrict__ Wbf, int n) {
  int e = blockIdx.x * blockDim.x + threadIdx.x;
  if (e < n) Wbf[e] = (__bf16)w[e];
}

// ---------------------------------------------------------------------------
// 4) Per-window LVC biases:  biasb[b][layer][l][o]
// ---------------------------------------------------------------------------
__global__ void k_biasb(const float* __restrict__ hkp, const float* __restrict__ w,
                        const float* __restrict__ bias, float* __restrict__ out) {
  int e = blockIdx.x * blockDim.x + threadIdx.x;   // 4*4*512*64
  int o = e & 63; int l = (e >> 6) & 511; int il = (e >> 15) & 3; int b = e >> 17;
  int ch = il * 64 + o;
  float acc = bias[ch];
  for (int c = 0; c < 64; ++c) {
    const float* xp = hkp + (size_t)(b * 64 + c) * SLEN;
    const float* wp = w + (size_t)(ch * 64 + c) * 3;
#pragma unroll
    for (int kk = 0; kk < 3; ++kk) {
      int q = l + kk - 1;
      if (q >= 0 && q < SLEN) acc += xp[q] * wp[kk];
    }
  }
  out[e] = acc;
}

// ---------------------------------------------------------------------------
// 5) Heavy GEMM via v_wmma_f32_16x16x32_bf16 (one dilation layer):
//    D[6144 x 512] = W_layer[6144 x 192] @ B[192 x 512] per batch.
//    A and B rows are [.][K] contiguous -> each frag = 2x b128 loads.
//    Output scattered to [b][l][o][K96] bf16 (L2-resident 25MB chunk).
// ---------------------------------------------------------------------------
__global__ void k_kergemm(const __bf16* __restrict__ Wbf, const __bf16* __restrict__ Bbf,
                          const float* __restrict__ kbias, __bf16* __restrict__ ker,
                          int layer) {
  int lane = threadIdx.x & 31;
  int wave = threadIdx.x >> 5;
  int tile = blockIdx.x * 8 + wave;                // 4 * 384 * 32 tiles
  int nt = tile & 31; tile >>= 5;
  int mt = tile % 384; int b = tile / 384;
  int half = lane >> 4;
  int mrow = mt * 16 + (lane & 15);                // local row in [0,6144)
  const __bf16* Arow = Wbf + (size_t)(layer * 6144 + mrow) * 192;
  const __bf16* Brow = Bbf + ((size_t)b * SLEN + nt * 16 + (lane & 15)) * 192;
  v8f c = {0.f, 0.f, 0.f, 0.f, 0.f, 0.f, 0.f, 0.f};
#pragma unroll
  for (int ks = 0; ks < 6; ++ks) {
    int k0 = ks * 32;
    Frag16 a, bb;
    a.q[0]  = *(const uq4*)(Arow + k0 + (half << 3));
    a.q[1]  = *(const uq4*)(Arow + k0 + 16 + (half << 3));
    bb.q[0] = *(const uq4*)(Brow + k0 + (half << 3));
    bb.q[1] = *(const uq4*)(Brow + k0 + 16 + (half << 3));
    c = __builtin_amdgcn_wmma_f32_16x16x32_bf16(false, a.v, false, bb.v,
                                                (short)0, c, false, false);
  }
  int l = nt * 16 + (lane & 15);
#pragma unroll
  for (int r = 0; r < 8; ++r) {
    int row = mt * 16 + r + (half << 3);           // = (ci*64 + o)*3 + kk
    int ci = row / 192; int rem = row - ci * 192; int o = rem / 3; int kk = rem - o * 3;
    float v = c[r] + kbias[layer * 6144 + row];
    ker[(((size_t)b * SLEN + l) * 64 + o) * 96 + ci * 3 + kk] = (__bf16)v;
  }
}

// ---------------------------------------------------------------------------
// 6) Fused per-layer LVC. One 512-thread (16-wave) WG per (batch, window).
//    TDM tensor_load_to_lds streams the 12KB weight tile into LDS while
//    phase A computes the dilated conv window; s_wait_tensorcnt + barrier
//    hands it to the 16 per-wave WMMA tiles.
// ---------------------------------------------------------------------------
__global__ __launch_bounds__(512) void k_lvc(const float* __restrict__ hin,
                                             float* __restrict__ hout,
                                             const __bf16* __restrict__ ker,
                                             const float* __restrict__ biasbuf,
                                             const float* __restrict__ lvcw,
                                             const float* __restrict__ lvcb,
                                             int layer, int dil) {
  __shared__ __attribute__((aligned(16))) __bf16 xw[32 * 66];
  __shared__ __attribute__((aligned(16))) __bf16 patch[64 * 96];
  __shared__ __attribute__((aligned(16))) __bf16 kerS[64 * 96];
  __shared__ float tb[64 * 64];
  __shared__ float biasS[64];
  int tid = threadIdx.x;
  int l = blockIdx.x & 511;
  int b = blockIdx.x >> 9;
  const __bf16* kg = ker + (size_t)(b * SLEN + l) * 64 * 96;

#if defined(__has_builtin) && __has_builtin(__builtin_amdgcn_tensor_load_to_lds)
  // Async DMA of the 96x64 bf16 tile (1536 x 8B elements, 1-D) into kerS.
  if (tid < 32) {
    typedef unsigned int u32x4 __attribute__((ext_vector_type(4)));
    typedef int i32x4 __attribute__((ext_vector_type(4)));
    typedef int i32x8 __attribute__((ext_vector_type(8)));
    unsigned long long ga = (unsigned long long)(uintptr_t)kg;
    unsigned int ldsa = (unsigned int)(uintptr_t)(&kerS[0]);
    u32x4 g0 = { 1u,                                   // count=1 user descriptor
                 ldsa,                                  // lds_addr
                 (unsigned int)ga,                      // global_addr[31:0]
                 (unsigned int)((ga >> 32) & 0x01FFFFFFu) | (2u << 30) }; // [56:32]+type=2
    i32x8 g1 = { (int)(3u << 16),       // data_size = 8B
                 (int)(1536u << 16),    // tensor_dim0 = 1536 (low 16 @ bits63:48)
                 (int)(1u << 16),       // tensor_dim0 hi=0 | tensor_dim1 = 1
                 (int)(1536u << 16),    // tensor_dim1 hi=0 | tile_dim0 = 1536
                 0,                     // tile_dim1 = 0 (unused), tile_dim2 = 0
                 1536,                  // tensor_dim0_stride = 1536
                 0, 0 };
    i32x4 z4 = {0, 0, 0, 0};
#if __clang_major__ >= 23
    i32x8 z8 = {0, 0, 0, 0, 0, 0, 0, 0};
    __builtin_amdgcn_tensor_load_to_lds(g0, g1, z4, z4, z8, 0);
#else
    __builtin_amdgcn_tensor_load_to_lds(g0, g1, z4, z4, 0);
#endif
  }
#else
  if (tid < 96) __builtin_prefetch(kg + tid * 64, 0, 1);
#endif

  // phase A: dilated conv output for window samples p = l*64-1 .. l*64+64
  for (int idx = tid; idx < 32 * 66; idx += 512) {
    int c = idx / 66, j = idx - c * 66;
    int p = l * 64 - 1 + j;
    float v = 0.f;
    if (p >= 0 && p < TLEN) {
      float acc = lvcb[layer * 32 + c];
      const float* wp = lvcw + (size_t)(layer * 32 + c) * 32 * 3;
      for (int cc = 0; cc < 32; ++cc) {
        const float* xp = hin + ((size_t)(b * 32 + cc) << 15);
#pragma unroll
        for (int kk = 0; kk < 3; ++kk) {
          int q = p + (kk - 1) * dil;
          if (q >= 0 && q < TLEN) acc += lrelu(xp[q], 0.2f) * wp[cc * 3 + kk];
        }
      }
      v = lrelu(acc, 0.2f);
    }
    xw[idx] = (__bf16)v;
  }
  __syncthreads();

  // phase A2: expand taps into WMMA-friendly patch[s][K=ci*3+kk]
  for (int idx = tid; idx < 64 * 96; idx += 512) {
    int s = idx / 96, K = idx - s * 96;
    int ci = K / 3, kk = K - ci * 3;
    patch[idx] = xw[ci * 66 + s + kk];
  }
#if !(defined(__has_builtin) && __has_builtin(__builtin_amdgcn_tensor_load_to_lds))
  for (int idx = tid; idx < (96 * 64 * 2) / 16; idx += 512)
    ((uq4*)kerS)[idx] = ((const uq4*)kg)[idx];
#endif
  if (tid < 64) biasS[tid] = biasbuf[(((size_t)b * 4 + layer) * SLEN + l) * 64 + tid];
#if defined(__has_builtin) && __has_builtin(__builtin_amdgcn_tensor_load_to_lds)
  __builtin_amdgcn_s_wait_tensorcnt(0);
#endif
  __syncthreads();

  // phase C: 16 waves, each one 16x16 tile of D[s=64][o=64], K=96 in 3 steps;
  // every fragment is two ds_load_b128s.
  {
    int lane = tid & 31;
    int wv = tid >> 5;
    int mtile = wv & 3, ntile = wv >> 2;
    int half = lane >> 4;
    int sl = mtile * 16 + (lane & 15);
    int o  = ntile * 16 + (lane & 15);
    const __bf16* Ar = patch + sl * 96;
    const __bf16* Br = kerS + o * 96;
    v8f c = {0.f, 0.f, 0.f, 0.f, 0.f, 0.f, 0.f, 0.f};
#pragma unroll
    for (int ks = 0; ks < 3; ++ks) {
      int k0 = ks * 32;
      Frag16 a, bb;
      a.q[0]  = *(const uq4*)(Ar + k0 + (half << 3));
      a.q[1]  = *(const uq4*)(Ar + k0 + 16 + (half << 3));
      bb.q[0] = *(const uq4*)(Br + k0 + (half << 3));
      bb.q[1] = *(const uq4*)(Br + k0 + 16 + (half << 3));
      c = __builtin_amdgcn_wmma_f32_16x16x32_bf16(false, a.v, false, bb.v,
                                                  (short)0, c, false, false);
    }
#pragma unroll
    for (int r = 0; r < 8; ++r) {
      int srow = mtile * 16 + r + (half << 3);
      tb[srow * 64 + o] = c[r] + biasS[o];
    }
  }
  __syncthreads();

  // phase D: gate + residual, write h_out
  for (int idx = tid; idx < 64 * 32; idx += 512) {
    int s = idx >> 5, o = idx & 31;
    float t1 = tb[s * 64 + o];
    float t2 = tb[s * 64 + o + 32];
    float g = (1.f / (1.f + __expf(-t1))) * tanhf(t2);
    size_t pos = ((size_t)(b * 32 + o) << 15) + (size_t)l * 64 + s;
    hout[pos] = hin[pos] + g;
  }
}

// ---------------------------------------------------------------------------
extern "C" void kernel_launch(void* const* d_in, const int* in_sizes, int n_in,
                              void* d_out, int out_size, void* d_ws, size_t ws_size,
                              hipStream_t stream) {
  const float* hidden      = (const float*)d_in[0];
  const float* spec        = (const float*)d_in[1];
  const float* convt_w     = (const float*)d_in[2];
  const float* convt_b     = (const float*)d_in[3];
  const float* kp_in_w     = (const float*)d_in[4];
  const float* kp_in_b     = (const float*)d_in[5];
  const float* kp_res_w1   = (const float*)d_in[6];
  const float* kp_res_b1   = (const float*)d_in[7];
  const float* kp_res_w2   = (const float*)d_in[8];
  const float* kp_res_b2   = (const float*)d_in[9];
  const float* kp_kernel_w = (const float*)d_in[10];
  const float* kp_kernel_b = (const float*)d_in[11];
  const float* kp_bias_w   = (const float*)d_in[12];
  const float* kp_bias_b   = (const float*)d_in[13];
  const float* lvc_w       = (const float*)d_in[14];
  const float* lvc_b       = (const float*)d_in[15];
  float* out = (float*)d_out;
  (void)in_sizes; (void)n_in; (void)out_size; (void)ws_size;

  char* p = (char*)d_ws;
  auto wsalloc = [&](size_t bytes) {
    char* r = p; p += (bytes + 255) & ~(size_t)255; return r;
  };
  float*  hbuf  = (float*)wsalloc((size_t)NB * 32 * TLEN * 4);   // 16 MB ping buffer
  float*  hkpA  = (float*)wsalloc((size_t)NB * 64 * SLEN * 4);
  float*  hkpB  = (float*)wsalloc((size_t)NB * 64 * SLEN * 4);
  __bf16* Bbf   = (__bf16*)wsalloc((size_t)NB * SLEN * 192 * 2);
  __bf16* Wbf   = (__bf16*)wsalloc((size_t)24576 * 192 * 2);
  __bf16* kerb  = (__bf16*)wsalloc((size_t)NB * SLEN * 64 * 96 * 2); // one layer, 25MB
  float*  biasb = (float*)wsalloc((size_t)NB * 4 * SLEN * 64 * 4);

  // 1) h = convT(lrelu(hidden))  -> d_out (ping-pong start)
  k_convt<<<16384, 256, 0, stream>>>(hidden, convt_w, convt_b, out);

  // 2) kernel-predictor frontend -> hkpA
  k_kp_in<<<512, 256, 0, stream>>>(spec, kp_in_w, kp_in_b, hkpA);
  for (int i = 0; i < 3; ++i) {
    k_kp_res<<<512, 256, 0, stream>>>(hkpA, kp_res_w1 + (size_t)i * 64 * 64 * 3,
                                      kp_res_b1 + i * 64, nullptr, hkpB);
    k_kp_res<<<512, 256, 0, stream>>>(hkpB, kp_res_w2 + (size_t)i * 64 * 64 * 3,
                                      kp_res_b2 + i * 64, hkpA, hkpA);
  }

  // 3) GEMM operand prep (bf16)
  k_im2col<<<1536, 256, 0, stream>>>(hkpA, Bbf);
  k_wcvt<<<18432, 256, 0, stream>>>(kp_kernel_w, Wbf, 24576 * 192);

  // 4) per-window biases
  k_biasb<<<2048, 256, 0, stream>>>(hkpA, kp_bias_w, kp_bias_b, biasb);

  // 5) four LVC layers; per layer: WMMA kernel-predictor GEMM (25MB chunk,
  //    L2-resident) immediately consumed by the fused WMMA LVC kernel.
  const int dil[4] = {1, 3, 9, 27};
  float* hcur = out;   // holds current h
  float* halt = hbuf;
  for (int i = 0; i < 4; ++i) {
    k_kergemm<<<6144, 256, 0, stream>>>(Wbf, Bbf, kp_kernel_b, kerb, i);
    k_lvc<<<2048, 512, 0, stream>>>(hcur, halt, kerb, biasb, lvc_w, lvc_b, i, dil[i]);
    float* t = hcur; hcur = halt; halt = t;
  }
  // after 4 swaps the final write landed in d_out and hcur == out
}
